// SimpleAttentiveFP_65524021067921
// MI455X (gfx1250) — matrix operand order repaired and checked by reading, not compiled, not run on hardware
//
#include <hip/hip_runtime.h>
#include <math.h>

// ---------------- problem constants ----------------
#define Hh   128
#define H3   384
#define FE   16
#define FN   64
#define SLOPEF 0.01f

typedef __attribute__((ext_vector_type(16))) _Float16 v16h;
typedef __attribute__((ext_vector_type(8)))  _Float16 h8;
typedef __attribute__((ext_vector_type(4)))  _Float16 h4;
typedef __attribute__((ext_vector_type(8)))  float    v8f;

union V16U { v16h v; h8 h[2]; };

__device__ __forceinline__ float leakyf(float v) { return v >= 0.f ? v : SLOPEF * v; }
__device__ __forceinline__ float sigmf(float v)  { return 1.f / (1.f + expf(-v)); }

__device__ __forceinline__ float atomicMaxFloat(float* addr, float val) {
  if (val >= 0.f)
    return __int_as_float(atomicMax((int*)addr, __float_as_int(val)));
  return __uint_as_float(atomicMin((unsigned int*)addr, __float_as_uint(val)));
}

// CDNA5 WMMA: D = A(16x32 f16) x B(32x16 f16) + C(16x16 f32)
__device__ __forceinline__ v8f wmma16x16x32(v16h a, v16h b, v8f c) {
  return __builtin_amdgcn_wmma_f32_16x16x32_f16(false, a, false, b, (short)0, c, false, false);
}

// ---- vectorized f32 -> f16 LDS staging helpers (16B global loads, 16B LDS stores) ----
__device__ __forceinline__ void pack4(const float* __restrict__ g, _Float16* dst) {
  float4 f = *(const float4*)g;
  h4 a; a[0] = (_Float16)f.x; a[1] = (_Float16)f.y; a[2] = (_Float16)f.z; a[3] = (_Float16)f.w;
  *(h4*)dst = a;
}
__device__ __forceinline__ void pack8(const float* __restrict__ g, _Float16* dst) {
  float4 f0 = ((const float4*)g)[0], f1 = ((const float4*)g)[1];
  h8 a;
  a[0] = (_Float16)f0.x; a[1] = (_Float16)f0.y; a[2] = (_Float16)f0.z; a[3] = (_Float16)f0.w;
  a[4] = (_Float16)f1.x; a[5] = (_Float16)f1.y; a[6] = (_Float16)f1.z; a[7] = (_Float16)f1.w;
  *(h8*)dst = a;
}
__device__ __forceinline__ void pack16(const float* __restrict__ g, _Float16* dst) {
  pack8(g, dst); pack8(g + 8, dst + 8);
}
__device__ __forceinline__ void zero4(_Float16* dst)  { h4 z = {}; *(h4*)dst = z; }
__device__ __forceinline__ void zero16(_Float16* dst) { h8 z = {}; ((h8*)dst)[0] = z; ((h8*)dst)[1] = z; }

// A fragment (16x32 f16), As laid out [row][32k]:
// lane L holds row L%16; element e -> k = 16*(e>=8) + 8*(L>=16) + (e&7)
// => two contiguous 8-half (16B) runs at k = kh and k = 16+kh.
__device__ __forceinline__ v16h load_a_frag(const _Float16* As, int row, int lane) {
  int kh = (lane >> 4) << 3;
  V16U u;
  u.h[0] = *(const h8*)&As[row * 32 + kh];
  u.h[1] = *(const h8*)&As[row * 32 + 16 + kh];
  return u.v;
}
// B fragment (32x16 f16), Bst laid out [col][32k]:
// lane L holds col L%16; element e -> k = 16*(e>=8) + 8*(L>=16) + (e&7)  => two 8-half runs.
__device__ __forceinline__ v16h load_b_frag(const _Float16* Bst, int col, int lane) {
  int kb = (lane >> 4) << 3;
  V16U u;
  u.h[0] = *(const h8*)&Bst[col * 32 + kb];
  u.h[1] = *(const h8*)&Bst[col * 32 + 16 + kb];
  return u.v;
}

// ---------------- WMMA GEMM: Y[M,NOUT] = act(X[M,K] @ W^T + bias) ----------------
// W row-major [NOUT,K] fp32.  K % 32 == 0, NOUT % 64 == 0.  act: 0=none, 1=leaky.
// Tile 128x64, 8 waves; each wave: 16 rows x 64 cols = 4 WMMA per K-step (A reuse x4).
__global__ __launch_bounds__(256) void gemm_xwT_kernel(
    const float* __restrict__ X, const float* __restrict__ Wt,
    const float* __restrict__ bias, float* __restrict__ Y,
    int M, int K, int NOUT, int act)
{
  __shared__ _Float16 As[128 * 32];   // [row][k]
  __shared__ _Float16 Bs[64 * 32];    // [col][k]
  int tid = threadIdx.x, lane = tid & 31, wave = tid >> 5;
  int rowBase = blockIdx.y * 128;
  int colBase = blockIdx.x * 64;

  // staging ownership: A: thread -> row tid/2, 16 k's; B: thread -> col tid/4, 8 k's
  int ar = tid >> 1, ak = (tid & 1) * 16;
  int bc = tid >> 2, bk = (tid & 3) * 8;
  int arow = rowBase + ar;
  bool avalid = arow < M;
  const float* aptr = X + (size_t)arow * K + ak;
  const float* bptr = Wt + (size_t)(colBase + bc) * K + bk;
  if (!avalid) zero16(&As[ar * 32 + ak]);   // zero once; never rewritten

  v8f acc[4] = {};
  for (int kb = 0; kb < K; kb += 32) {
    if (avalid) pack16(aptr + kb, &As[ar * 32 + ak]);
    pack8(bptr + kb, &Bs[bc * 32 + bk]);
    __syncthreads();
    v16h a = load_a_frag(As, wave * 16 + (lane & 15), lane);
#pragma unroll
    for (int j = 0; j < 4; ++j) {
      v16h b = load_b_frag(Bs, j * 16 + (lane & 15), lane);
      acc[j] = wmma16x16x32(a, b, acc[j]);
    }
    __syncthreads();
  }
  int colb = colBase + (lane & 15);
  float bv[4];
#pragma unroll
  for (int j = 0; j < 4; ++j) bv[j] = bias ? bias[colb + j * 16] : 0.f;
  int rbase = rowBase + wave * 16 + ((lane >> 4) << 3);
#pragma unroll
  for (int r = 0; r < 8; ++r) {
    int row = rbase + r;
    if (row < M) {
#pragma unroll
      for (int j = 0; j < 4; ++j) {
        float v = acc[j][r] + bv[j];
        if (act == 1) v = leakyf(v);
        Y[(size_t)row * NOUT + colb + j * 16] = v;
      }
    }
  }
}

// ---------------- fused GATEConv edge-attention logit kernel ----------------
// Per edge e: hj = leaky([x[src]||edge_attr] @ W1^T) (never stored),
//             alpha[e] = leaky(hj . att_l + xr[dst]);  atomicMax m[dst].
__global__ __launch_bounds__(256) void gate_alpha_kernel(
    const float* __restrict__ xh, const float* __restrict__ eattr,
    const int* __restrict__ src, const int* __restrict__ dst,
    const float* __restrict__ W1 /*[128,144]*/, const float* __restrict__ att_l,
    const float* __restrict__ xr /*[N]*/,
    float* __restrict__ alpha, float* __restrict__ m, int E)
{
  __shared__ _Float16 Ae[32 * 32];    // [edge][k]
  __shared__ _Float16 Bw[128 * 32];   // [col][k]
  __shared__ float partial[32];
  __shared__ int sidx[32], didx[32];
  int tid = threadIdx.x, lane = tid & 31, wave = tid >> 5;
  int eBase = blockIdx.x * 32;
  if (tid < 32) {
    int e = eBase + tid;
    int sv = 0, dv = 0;
    if (e < E) { sv = src[e]; dv = dst[e]; }
    sidx[tid] = sv; didx[tid] = dv;
    partial[tid] = 0.f;
  }
  __syncthreads();

  // staging ownership: A: thread -> edge-row tid/8, 4 k's; B: thread -> col tid/2, 16 k's
  int ar = tid >> 3, ak = (tid & 7) * 4;
  int bc = tid >> 1, bk = (tid & 1) * 16;
  bool ev = (eBase + ar) < E;
  const float* xrow = xh + (size_t)sidx[ar] * Hh;
  const float* erow = eattr + (size_t)(eBase + ar) * FE;
  const float* wrow = W1 + (size_t)bc * (Hh + FE);
  if (!ev) zero4(&Ae[ar * 32 + ak]);  // zero once

  int wr = wave & 1;    // 2 row-groups of 16 edges
  int wc = wave >> 1;   // 4 col-groups of 32 outputs
  v8f acc0 = {}; v8f acc1 = {};
  for (int kc = 0; kc < 5; ++kc) {    // K = 144 padded to 160
    int kg0 = kc * 32;
    if (ev) {
      int kg = kg0 + ak;              // chunks of 4 never straddle 128/144 boundaries
      if (kg < Hh)           pack4(xrow + kg, &Ae[ar * 32 + ak]);
      else if (kg < Hh + FE) pack4(erow + (kg - Hh), &Ae[ar * 32 + ak]);
      else                   zero4(&Ae[ar * 32 + ak]);
    }
    {
      int kg = kg0 + bk;              // 16-chunks align with the 144 boundary
      if (kg < Hh + FE) pack16(wrow + kg, &Bw[bc * 32 + bk]);
      else              zero16(&Bw[bc * 32 + bk]);
    }
    __syncthreads();
    v16h a = load_a_frag(Ae, wr * 16 + (lane & 15), lane);
    int colB = wc * 32 + (lane & 15);
    v16h b0 = load_b_frag(Bw, colB, lane);
    v16h b1 = load_b_frag(Bw, colB + 16, lane);
    acc0 = wmma16x16x32(a, b0, acc0);
    acc1 = wmma16x16x32(a, b1, acc1);
    __syncthreads();
  }
  { // epilogue: leaky(hj) . att_l reduced per edge via LDS atomics
    int colA = wc * 32 + (lane & 15);
    float wA = att_l[colA], wB = att_l[colA + 16];
    int rbase = wr * 16 + ((lane >> 4) << 3);
#pragma unroll
    for (int r = 0; r < 8; ++r) {
      atomicAdd(&partial[rbase + r], leakyf(acc0[r]) * wA + leakyf(acc1[r]) * wB);
    }
  }
  __syncthreads();
  if (tid < 32) {
    int e = eBase + tid;
    if (e < E) {
      float al = leakyf(partial[tid] + xr[didx[tid]]);
      alpha[e] = al;
      atomicMaxFloat(&m[didx[tid]], al);
    }
  }
}

// ---------------- elementwise / scatter helpers ----------------
__global__ void fill_kernel(float* __restrict__ p, float v, long long n) {
  long long i = (long long)blockIdx.x * blockDim.x + threadIdx.x;
  if (i < n) p[i] = v;
}

__global__ void rowdot_kernel(const float* __restrict__ X, const float* __restrict__ v,
                              float* __restrict__ out, int M, int K) {
  int i = blockIdx.x * blockDim.x + threadIdx.x;
  if (i >= M) return;
  float s = 0.f;
  for (int k = 0; k < K; ++k) s += X[(size_t)i * K + k] * v[k];
  out[i] = s;
}

__global__ void gat_alpha_kernel(const float* __restrict__ asrc, const float* __restrict__ adst,
                                 const int* __restrict__ src, const int* __restrict__ dst,
                                 float* __restrict__ alpha, float* __restrict__ m, int E) {
  int e = blockIdx.x * blockDim.x + threadIdx.x;
  if (e >= E) return;
  float al = leakyf(asrc[src[e]] + adst[dst[e]]);
  alpha[e] = al;
  atomicMaxFloat(&m[dst[e]], al);
}

__global__ void mol_alpha_kernel(const float* __restrict__ asrcN, const float* __restrict__ adstG,
                                 const int* __restrict__ batch,
                                 float* __restrict__ alphaN, float* __restrict__ m, int Nn) {
  int n = blockIdx.x * blockDim.x + threadIdx.x;
  if (n >= Nn) return;
  int b = batch[n];
  float al = leakyf(asrcN[n] + adstG[b]);
  alphaN[n] = al;
  atomicMaxFloat(&m[b], al);
}

__global__ void seg_exp_kernel(float* __restrict__ a, const float* __restrict__ m,
                               float* __restrict__ s, const int* __restrict__ seg, int n) {
  int i = blockIdx.x * blockDim.x + threadIdx.x;
  if (i >= n) return;
  int b = seg[i];
  float v = expf(a[i] - m[b]);
  a[i] = v;
  atomicAdd(&s[b], v);
}

// out[seg[i]] += feat[gather? gather[i] : i] * alpha[i]/(s[seg[i]]+eps), 128 lanes/row
__global__ void scatter_wsum_kernel(const float* __restrict__ alpha, const float* __restrict__ s,
                                    const float* __restrict__ feat, const int* __restrict__ gather,
                                    const int* __restrict__ seg, float* __restrict__ out, long long n) {
  long long idx = (long long)blockIdx.x * blockDim.x + threadIdx.x;
  if (idx >= n * Hh) return;
  int i = (int)(idx >> 7), c = (int)(idx & 127);
  int d = seg[i];
  float a = alpha[i] / (s[d] + 1e-16f);
  int r = gather ? gather[i] : i;
  atomicAdd(&out[(size_t)d * Hh + c], feat[(size_t)r * Hh + c] * a);
}

__global__ void pool_kernel(const float* __restrict__ x, const int* __restrict__ batch,
                            float* __restrict__ outG, long long Nn) {
  long long idx = (long long)blockIdx.x * blockDim.x + threadIdx.x;
  if (idx >= Nn * Hh) return;
  int n = (int)(idx >> 7), c = (int)(idx & 127);
  atomicAdd(&outG[(size_t)batch[n] * Hh + c], x[idx]);
}

__global__ void relu_kernel(float* __restrict__ p, long long n) {
  long long i = (long long)blockIdx.x * blockDim.x + threadIdx.x;
  if (i < n) { float v = p[i]; p[i] = v > 0.f ? v : 0.f; }
}

__global__ void elu_bias_kernel(float* __restrict__ a, const float* __restrict__ bias, long long rows) {
  long long idx = (long long)blockIdx.x * blockDim.x + threadIdx.x;
  if (idx >= rows * Hh) return;
  int c = (int)(idx & 127);
  float v = a[idx] + bias[c];
  a[idx] = v > 0.f ? v : (expf(v) - 1.f);
}

// GRUCell gate order (r,z,n); state <- relu((1-z)*tanh(in+r*hn) + z*state)
__global__ void gru_combine_kernel(const float* __restrict__ gi, const float* __restrict__ gh,
                                   float* __restrict__ state, long long rows) {
  long long idx = (long long)blockIdx.x * blockDim.x + threadIdx.x;
  if (idx >= rows * Hh) return;
  int n = (int)(idx >> 7), c = (int)(idx & 127);
  size_t b = (size_t)n * H3;
  float r  = sigmf(gi[b + c]        + gh[b + c]);
  float z  = sigmf(gi[b + Hh + c]   + gh[b + Hh + c]);
  float ng = tanhf(gi[b + 2*Hh + c] + r * gh[b + 2*Hh + c]);
  float o  = (1.f - z) * ng + z * state[idx];
  state[idx] = o > 0.f ? o : 0.f;
}

__global__ void final_kernel(const float* __restrict__ out2, const float* __restrict__ finW,
                             const float* __restrict__ finb, float* __restrict__ y, int Gn) {
  int g = blockIdx.x * blockDim.x + threadIdx.x;
  if (g >= Gn) return;
  float s = finb[0];
  for (int c = 0; c < Hh; ++c) s += out2[(size_t)g * Hh + c] * finW[c];
  y[g] = 1.f / (1.f + expf(-s));
}

// ---------------- driver ----------------
extern "C" void kernel_launch(void* const* d_in, const int* in_sizes, int n_in,
                              void* d_out, int out_size, void* d_ws, size_t ws_size,
                              hipStream_t stream) {
  const float* x_in  = (const float*)d_in[0];
  const int*   eidx  = (const int*)  d_in[1];
  const float* eattr = (const float*)d_in[2];
  const int*   batch = (const int*)  d_in[3];
  const float* lin1_W = (const float*)d_in[4];
  const float* lin1_b = (const float*)d_in[5];
  const float* gate_att_l = (const float*)d_in[6];
  const float* gate_att_r = (const float*)d_in[7];
  const float* gate_lin1_W = (const float*)d_in[8];
  const float* gate_lin2_W = (const float*)d_in[9];
  const float* gate_bias   = (const float*)d_in[10];
  const float* gru0_Wih = (const float*)d_in[11];
  const float* gru0_Whh = (const float*)d_in[12];
  const float* gru0_bih = (const float*)d_in[13];
  const float* gru0_bhh = (const float*)d_in[14];
  const float* atom_lin_W   = (const float*)d_in[15];
  const float* atom_att_src = (const float*)d_in[16];
  const float* atom_att_dst = (const float*)d_in[17];
  const float* atom_bias    = (const float*)d_in[18];
  const float* atom_gru_Wih = (const float*)d_in[19];
  const float* atom_gru_Whh = (const float*)d_in[20];
  const float* atom_gru_bih = (const float*)d_in[21];
  const float* atom_gru_bhh = (const float*)d_in[22];
  const float* mol_lin_W   = (const float*)d_in[23];
  const float* mol_att_src = (const float*)d_in[24];
  const float* mol_att_dst = (const float*)d_in[25];
  const float* mol_bias    = (const float*)d_in[26];
  const float* mol_gru_Wih = (const float*)d_in[27];
  const float* mol_gru_Whh = (const float*)d_in[28];
  const float* mol_gru_bih = (const float*)d_in[29];
  const float* mol_gru_bhh = (const float*)d_in[30];
  const float* lin2_W = (const float*)d_in[31];
  const float* lin2_b = (const float*)d_in[32];
  const float* fin_W  = (const float*)d_in[33];
  const float* fin_b  = (const float*)d_in[34];

  const int N = in_sizes[0] / FN;
  const int E = in_sizes[2] / FE;
  const int G = out_size;
  const int* src = eidx;
  const int* dst = eidx + E;

  float* ws = (float*)d_ws;
  size_t o = 0;
  auto alloc = [&](size_t n) { float* p = ws + o; o += n; return p; };
  float* xh    = alloc((size_t)N * Hh);   // node state
  float* xs    = alloc((size_t)N * Hh);   // per-layer projected sources / messages
  float* agg   = alloc((size_t)N * Hh);   // segment-sum accumulator (reused as hmsg)
  float* gi    = alloc((size_t)N * H3);
  float* gh    = alloc((size_t)N * H3);
  float* alpha = alloc((size_t)(E > N ? E : N));
  float* mbuf  = alloc((size_t)N);
  float* sbuf  = alloc((size_t)N);
  float* asrc  = alloc((size_t)N);
  float* adst  = alloc((size_t)N);
  float* outG  = alloc((size_t)G * Hh);
  float* odG   = alloc((size_t)G * Hh);
  float* hgG   = alloc((size_t)G * Hh);
  float* giG   = alloc((size_t)G * H3);
  float* ghG   = alloc((size_t)G * H3);
  float* out2  = alloc((size_t)G * Hh);
  (void)ws_size; (void)n_in;

  dim3 blk(256);
  auto g1 = [](long long n) { return dim3((unsigned)((n + 255) / 256)); };
  auto gg = [](int M, int NOUT) { return dim3((unsigned)(NOUT / 64), (unsigned)((M + 127) / 128)); };
  const long long NH = (long long)N * Hh, EH = (long long)E * Hh, GH = (long long)G * Hh;

  // ---- input projection: x = leaky(x_in @ lin1_W^T + b) ----
  gemm_xwT_kernel<<<gg(N, Hh), blk, 0, stream>>>(x_in, lin1_W, lin1_b, xh, N, FN, Hh, 1);

  // ---- GATEConv ----
  rowdot_kernel<<<g1(N), blk, 0, stream>>>(xh, gate_att_r, asrc, N, Hh);                 // xr[dst]
  gemm_xwT_kernel<<<gg(N, Hh), blk, 0, stream>>>(xh, gate_lin2_W, nullptr, xs, N, Hh, Hh, 0);
  fill_kernel<<<g1(N), blk, 0, stream>>>(mbuf, -1e30f, N);
  fill_kernel<<<g1(N), blk, 0, stream>>>(sbuf, 0.f, N);
  fill_kernel<<<g1(NH), blk, 0, stream>>>(agg, 0.f, NH);
  gate_alpha_kernel<<<dim3((unsigned)((E + 31) / 32)), blk, 0, stream>>>(
      xh, eattr, src, dst, gate_lin1_W, gate_att_l, asrc, alpha, mbuf, E);
  seg_exp_kernel<<<g1(E), blk, 0, stream>>>(alpha, mbuf, sbuf, dst, E);
  scatter_wsum_kernel<<<g1(EH), blk, 0, stream>>>(alpha, sbuf, xs, src, dst, agg, E);
  elu_bias_kernel<<<g1(NH), blk, 0, stream>>>(agg, gate_bias, N);
  gemm_xwT_kernel<<<gg(N, H3), blk, 0, stream>>>(agg, gru0_Wih, gru0_bih, gi, N, Hh, H3, 0);
  gemm_xwT_kernel<<<gg(N, H3), blk, 0, stream>>>(xh, gru0_Whh, gru0_bhh, gh, N, Hh, H3, 0);
  gru_combine_kernel<<<g1(NH), blk, 0, stream>>>(gi, gh, xh, N);

  // ---- atom-level GATConv layers ----
  for (int l = 0; l < 2; ++l) {
    gemm_xwT_kernel<<<gg(N, Hh), blk, 0, stream>>>(xh, atom_lin_W + (size_t)l * Hh * Hh,
                                                   nullptr, xs, N, Hh, Hh, 0);
    rowdot_kernel<<<g1(N), blk, 0, stream>>>(xs, atom_att_src + l * Hh, asrc, N, Hh);
    rowdot_kernel<<<g1(N), blk, 0, stream>>>(xs, atom_att_dst + l * Hh, adst, N, Hh);
    fill_kernel<<<g1(N), blk, 0, stream>>>(mbuf, -1e30f, N);
    fill_kernel<<<g1(N), blk, 0, stream>>>(sbuf, 0.f, N);
    fill_kernel<<<g1(NH), blk, 0, stream>>>(agg, 0.f, NH);
    gat_alpha_kernel<<<g1(E), blk, 0, stream>>>(asrc, adst, src, dst, alpha, mbuf, E);
    seg_exp_kernel<<<g1(E), blk, 0, stream>>>(alpha, mbuf, sbuf, dst, E);
    scatter_wsum_kernel<<<g1(EH), blk, 0, stream>>>(alpha, sbuf, xs, src, dst, agg, E);
    elu_bias_kernel<<<g1(NH), blk, 0, stream>>>(agg, atom_bias + l * Hh, N);
    gemm_xwT_kernel<<<gg(N, H3), blk, 0, stream>>>(agg, atom_gru_Wih + (size_t)l * H3 * Hh,
                                                   atom_gru_bih + l * H3, gi, N, Hh, H3, 0);
    gemm_xwT_kernel<<<gg(N, H3), blk, 0, stream>>>(xh, atom_gru_Whh + (size_t)l * H3 * Hh,
                                                   atom_gru_bhh + l * H3, gh, N, Hh, H3, 0);
    gru_combine_kernel<<<g1(NH), blk, 0, stream>>>(gi, gh, xh, N);
  }

  // ---- molecule readout ----
  fill_kernel<<<g1(GH), blk, 0, stream>>>(outG, 0.f, GH);
  pool_kernel<<<g1(NH), blk, 0, stream>>>(xh, batch, outG, N);
  relu_kernel<<<g1(GH), blk, 0, stream>>>(outG, GH);
  gemm_xwT_kernel<<<gg(N, Hh), blk, 0, stream>>>(xh, mol_lin_W, nullptr, xs, N, Hh, Hh, 0);
  rowdot_kernel<<<g1(N), blk, 0, stream>>>(xs, mol_att_src, asrc, N, Hh);
  for (int t = 0; t < 3; ++t) {
    gemm_xwT_kernel<<<gg(G, Hh), blk, 0, stream>>>(outG, mol_lin_W, nullptr, odG, G, Hh, Hh, 0);
    rowdot_kernel<<<g1(G), blk, 0, stream>>>(odG, mol_att_dst, adst, G, Hh);
    fill_kernel<<<g1(G), blk, 0, stream>>>(mbuf, -1e30f, G);
    fill_kernel<<<g1(G), blk, 0, stream>>>(sbuf, 0.f, G);
    fill_kernel<<<g1(GH), blk, 0, stream>>>(hgG, 0.f, GH);
    mol_alpha_kernel<<<g1(N), blk, 0, stream>>>(asrc, adst, batch, alpha, mbuf, N);
    seg_exp_kernel<<<g1(N), blk, 0, stream>>>(alpha, mbuf, sbuf, batch, N);
    scatter_wsum_kernel<<<g1(NH), blk, 0, stream>>>(alpha, sbuf, xs, nullptr, batch, hgG, N);
    elu_bias_kernel<<<g1(GH), blk, 0, stream>>>(hgG, mol_bias, G);
    gemm_xwT_kernel<<<gg(G, H3), blk, 0, stream>>>(hgG, mol_gru_Wih, mol_gru_bih, giG, G, Hh, H3, 0);
    gemm_xwT_kernel<<<gg(G, H3), blk, 0, stream>>>(outG, mol_gru_Whh, mol_gru_bhh, ghG, G, Hh, H3, 0);
    gru_combine_kernel<<<g1(GH), blk, 0, stream>>>(giG, ghG, outG, G);
  }

  // ---- final linears + sigmoid ----
  gemm_xwT_kernel<<<gg(G, Hh), blk, 0, stream>>>(outG, lin2_W, lin2_b, out2, G, Hh, Hh, 0);
  final_kernel<<<g1(G), blk, 0, stream>>>(out2, fin_W, fin_b, (float*)d_out, G);
}
